// DynamicBlock_57629871178610
// MI455X (gfx1250) — matrix operand order repaired and checked by reading, not compiled, not run on hardware
//
#include <hip/hip_runtime.h>

#define HW   65536
#define Hh   256
#define Ww   256
#define Cc   64
#define C4v  16
#define NP   262144   // B*H*W total pixels

typedef __attribute__((ext_vector_type(16))) _Float16 v16h;
typedef __attribute__((ext_vector_type(8)))  _Float16 v8h;
typedef __attribute__((ext_vector_type(8)))  float    v8f;

__device__ __forceinline__ v8f wmma_f16(v16h a, v16h b, v8f c) {
  // D = A(16x32 f16) * B(32x16 f16) + C(16x16 f32)
  return __builtin_amdgcn_wmma_f32_16x16x32_f16(false, a, false, b, (short)0, c,
                                                false, false);
}

// load 16 consecutive f32 and convert to the 16 halves of a B fragment
__device__ __forceinline__ void ld16h(v16h& dst, const float* p) {
  const float4* w4 = (const float4*)p;
#pragma unroll
  for (int q = 0; q < 4; ++q) {
    float4 f = w4[q];
    dst[q * 4 + 0] = (_Float16)f.x;
    dst[q * 4 + 1] = (_Float16)f.y;
    dst[q * 4 + 2] = (_Float16)f.z;
    dst[q * 4 + 3] = (_Float16)f.w;
  }
}

// Build the two K-step A fragments (K=64) for a 16-pixel M-tile directly from
// planar f32 global memory. Contiguous layout: lane<16 holds row r, K = k0..k0+15
// with k0=0; lanes 16-31 the same rows with k0=16. For each
// fixed K the 16 lanes of a half-wave read 16 consecutive pixels (coalesced),
// and the 32 lanes' fragments exactly partition the 16x64 tile (no dup).
__device__ __forceinline__ void ldA64(v16h& a0, v16h& a1, const float* basep,
                                      int k0) {
#pragma unroll
  for (int j = 0; j < 8; ++j) {
    a0[j]     = (_Float16)basep[(size_t)(k0 + j) * HW];
    a0[j + 8] = (_Float16)basep[(size_t)(k0 + 8 + j) * HW];
    a1[j]     = (_Float16)basep[(size_t)(k0 + 32 + j) * HW];
    a1[j + 8] = (_Float16)basep[(size_t)(k0 + 40 + j) * HW];
  }
}

// ---------------------------------------------------------------------------
// Bottleneck: t = relu(BN(in @ w1^T + b1)), t stored row-major [P][16] f16.
// GEMM: M=P, N=16, K=64 (2 WMMA K-steps). 8 waves/WG, 16 pixels per wave.
// ---------------------------------------------------------------------------
__global__ __launch_bounds__(256) void k_bott(
    const float* __restrict__ in, const float* __restrict__ w1,
    const float* __restrict__ bias1, const float* __restrict__ gamma,
    const float* __restrict__ beta, const float* __restrict__ mean,
    const float* __restrict__ var, _Float16* __restrict__ tout) {
  const int wave = threadIdx.x >> 5, lane = threadIdx.x & 31;
  const int m0 = (blockIdx.x * 8 + wave) * 16;  // global pixel base
  const int b = m0 >> 16;
  const int p0 = m0 & (HW - 1);

  const int r = lane & 15;
  const int k0 = (lane < 16) ? 0 : 16;  // A-fragment K split (contiguous layout)
  v16h a0, a1;
  ldA64(a0, a1, in + (size_t)b * Cc * HW + p0 + r, k0);

  const int n = lane & 15;              // output channel (N)
  const int ks = (lane < 16) ? 0 : 16;  // B-fragment K range
  v16h b0, b1;
  ld16h(b0, w1 + n * Cc + ks);
  ld16h(b1, w1 + n * Cc + 32 + ks);

  v8f c = {};
  c = wmma_f16(a0, b0, c);
  c = wmma_f16(a1, b1, c);

  const float sc = gamma[n] * rsqrtf(var[n] + 1e-5f);
  const float sh = beta[n] - mean[n] * sc;
  const float bb = bias1[n];
  const int moff = (lane < 16) ? 0 : 8;
#pragma unroll
  for (int j = 0; j < 8; ++j) {
    float v = (c[j] + bb) * sc + sh;
    v = fmaxf(v, 0.0f);
    tout[(size_t)(m0 + moff + j) * C4v + n] = (_Float16)v;
  }
}

// ---------------------------------------------------------------------------
// Dynamic branch: wdyn = t @ w2^T + b2 (M=16 px, N=576, K=16 padded to 32),
// staged in LDS, then out[p,c] = sum_k wdyn[p, c*9+k] * patch3x3(xin)[p,c,k].
// One 16-pixel row-segment tile per workgroup; 36 WMMA N-tiles over 8 waves.
// ---------------------------------------------------------------------------
__global__ __launch_bounds__(256) void k_dyn(
    const _Float16* __restrict__ tin, const float* __restrict__ w2,
    const float* __restrict__ bias2, const float* __restrict__ xin,
    float* __restrict__ outp) {
  __shared__ float Wd[16][584];  // 16 px x 576 dyn-weights (+pad)
  const int wave = threadIdx.x >> 5, lane = threadIdx.x & 31;
  const int m0 = blockIdx.x * 16;
  const int b = m0 >> 16;
  const int p0 = m0 & (HW - 1);
  const int y = p0 >> 8, x0 = p0 & 255;
  const int r = lane & 15;
  const int n = lane & 15;
  const int moff = (lane < 16) ? 0 : 8;

  // A fragment from row-major t: real K=0..15, K=16..31 zero padding
  v16h a = {};
  {
    const _Float16* trow = tin + (size_t)(m0 + r) * C4v;
    if (lane < 16) a = *(const v16h*)trow;
  }

  __builtin_prefetch(xin + (size_t)b * Cc * HW + p0, 0, 1);

  for (int nt = wave; nt < 36; nt += 8) {
    const int o = nt * 16 + n;
    v16h bf = {};                       // lanes 16-31: K=16..31 => zero pad
    if (lane < 16) ld16h(bf, w2 + o * C4v);
    v8f d = {};
    d = wmma_f16(a, bf, d);
    const float bb = bias2[o];
#pragma unroll
    for (int j = 0; j < 8; ++j) Wd[moff + j][o] = d[j] + bb;
  }
  __syncthreads();

  // content-adaptive 3x3 depthwise reduction (F.unfold tap order)
  for (int q = threadIdx.x; q < 1024; q += 256) {
    const int c = q >> 4, rr = q & 15;
    const float* wp = &Wd[rr][c * 9];
    const float* inp = xin + ((size_t)b * Cc + c) * HW;
    const int xx = x0 + rr;
    float acc = 0.0f;
#pragma unroll
    for (int dy = -1; dy <= 1; ++dy) {
      const int yy = y + dy;
      if (yy < 0 || yy > Hh - 1) continue;
      const float* rowp = inp + yy * Ww;
#pragma unroll
      for (int dx = -1; dx <= 1; ++dx) {
        const int xc = xx + dx;
        if (xc < 0 || xc > Ww - 1) continue;
        acc += wp[(dy + 1) * 3 + (dx + 1)] * rowp[xc];
      }
    }
    outp[((size_t)b * Cc + c) * HW + p0 + rr] = acc;
  }
}

// ---------------------------------------------------------------------------
// Pointwise 64x64 GEMM. mode 0: +bias then leaky_relu(0.1) -> outp
//                       mode 1: +bias then +resid (residual) -> outp
// ---------------------------------------------------------------------------
__global__ __launch_bounds__(256) void k_pw(
    const float* __restrict__ in, const float* __restrict__ w,
    const float* __restrict__ bias, const float* __restrict__ resid,
    float* __restrict__ outp, int mode) {
  const int wave = threadIdx.x >> 5, lane = threadIdx.x & 31;
  const int m0 = (blockIdx.x * 8 + wave) * 16;
  const int b = m0 >> 16;
  const int p0 = m0 & (HW - 1);

  const int r = lane & 15;
  const int k0 = (lane < 16) ? 0 : 16;
  v16h a0, a1;
  ldA64(a0, a1, in + (size_t)b * Cc * HW + p0 + r, k0);

  const int n = lane & 15;
  const int ks = (lane < 16) ? 0 : 16;
  const int moff = (lane < 16) ? 0 : 8;
#pragma unroll
  for (int nt = 0; nt < 4; ++nt) {
    const int o = nt * 16 + n;
    v16h b0, b1;
    ld16h(b0, w + o * Cc + ks);
    ld16h(b1, w + o * Cc + 32 + ks);
    v8f d = {};
    d = wmma_f16(a0, b0, d);
    d = wmma_f16(a1, b1, d);
    const float bb = bias[o];
#pragma unroll
    for (int j = 0; j < 8; ++j) {
      float v = d[j] + bb;
      const size_t oidx = ((size_t)b * Cc + o) * HW + p0 + moff + j;
      if (mode == 0) {
        v = (v > 0.0f) ? v : 0.1f * v;
      } else {
        v += resid[oidx];
      }
      outp[oidx] = v;
    }
  }
}

extern "C" void kernel_launch(void* const* d_in, const int* in_sizes, int n_in,
                              void* d_out, int out_size, void* d_ws,
                              size_t ws_size, hipStream_t stream) {
  (void)in_sizes; (void)n_in; (void)out_size; (void)ws_size;

  const float* x       = (const float*)d_in[0];
  const float* k1_w1   = (const float*)d_in[1];
  const float* k1_b1   = (const float*)d_in[2];
  const float* k1_g    = (const float*)d_in[3];
  const float* k1_be   = (const float*)d_in[4];
  const float* k1_m    = (const float*)d_in[5];
  const float* k1_v    = (const float*)d_in[6];
  const float* k1_w2   = (const float*)d_in[7];
  const float* k1_b2   = (const float*)d_in[8];
  const float* p1_w    = (const float*)d_in[9];
  const float* p1_b    = (const float*)d_in[10];
  const float* k2_w1   = (const float*)d_in[11];
  const float* k2_b1   = (const float*)d_in[12];
  const float* k2_g    = (const float*)d_in[13];
  const float* k2_be   = (const float*)d_in[14];
  const float* k2_m    = (const float*)d_in[15];
  const float* k2_v    = (const float*)d_in[16];
  const float* k2_w2   = (const float*)d_in[17];
  const float* k2_b2   = (const float*)d_in[18];
  const float* p2_w    = (const float*)d_in[19];
  const float* p2_b    = (const float*)d_in[20];
  float* out = (float*)d_out;

  // workspace: t [P][16] f16 (8 MB) + two planar f32 activation buffers;
  // total ~142 MB -> entire inter-kernel working set is L2-resident (192 MB).
  _Float16* tbuf = (_Float16*)d_ws;
  float* buf1 = (float*)((char*)d_ws + (size_t)NP * C4v * sizeof(_Float16));
  float* buf2 = buf1 + (size_t)NP * Cc;

  const dim3 blk(256);
  const int g_gemm = NP / 128;  // 8 waves x 16 px per WG
  const int g_dyn  = NP / 16;   // 16 px per WG

  // block 1
  k_bott<<<g_gemm, blk, 0, stream>>>(x, k1_w1, k1_b1, k1_g, k1_be, k1_m, k1_v,
                                     tbuf);
  k_dyn<<<g_dyn, blk, 0, stream>>>(tbuf, k1_w2, k1_b2, x, buf1);
  k_pw<<<g_gemm, blk, 0, stream>>>(buf1, p1_w, p1_b, nullptr, buf2, 0);
  // block 2
  k_bott<<<g_gemm, blk, 0, stream>>>(buf2, k2_w1, k2_b1, k2_g, k2_be, k2_m,
                                     k2_v, tbuf);
  k_dyn<<<g_dyn, blk, 0, stream>>>(tbuf, k2_w2, k2_b2, buf2, buf1);
  // final pointwise + residual
  k_pw<<<g_gemm, blk, 0, stream>>>(buf1, p2_w, p2_b, x, out, 1);
}
